// SubspaceMatching_11871289606891
// MI455X (gfx1250) — matrix operand order repaired and checked by reading, not compiled, not run on hardware
//
#include <hip/hip_runtime.h>
#include <hip/hip_bf16.h>

// ---------------------------------------------------------------------------
// SubspaceMatching pipeline for MI455X (gfx1250, wave32, WMMA, async-LDS).
// B=2,S=4 -> BS=8 slices; M=N=2048; C=256; 3 Sinkhorn iters.
// ---------------------------------------------------------------------------

#define BSLICES   8
#define MDIM      2048
#define NDIM      2048
#define CDIM      256
#define SKH_ITERS 3
#define CONF_THR  0.2f
#define MASK_THR  0.3f
#define UVSTRIDE  2064          // 2049 rounded up, per-slice stride for u/v
#define NEG_INF   (-__builtin_inff())

// sim GEMM macro-tile
#define TILE_M    128
#define TILE_N    64
#define SMEM_A_BYTES (TILE_M * CDIM * 2)     // 64 KB
#define SMEM_B_BYTES (TILE_N * CDIM * 2)     // 32 KB
#define SMEM_BYTES   (SMEM_A_BYTES + SMEM_B_BYTES)

typedef _Float16 half_t;
typedef __attribute__((ext_vector_type(16))) _Float16 v16h;
typedef __attribute__((ext_vector_type(8)))  _Float16 v8h;
typedef __attribute__((ext_vector_type(8)))  float    v8f;
typedef __attribute__((ext_vector_type(4)))  float    v4f;

#if __has_builtin(__builtin_amdgcn_global_load_async_to_lds_b128) && \
    __has_builtin(__builtin_amdgcn_s_wait_asynccnt)
#define HAVE_ASYNC_LDS 1
#else
#define HAVE_ASYNC_LDS 0
#endif

#if HAVE_ASYNC_LDS
typedef int b128_t __attribute__((vector_size(16)));
typedef __attribute__((address_space(1))) b128_t* gptr128;
typedef __attribute__((address_space(3))) b128_t* lptr128;

__device__ __forceinline__ void async_ld_b128(const half_t* g, half_t* l) {
  b128_t* gg = (b128_t*)g;   // generic, const dropped by C cast
  b128_t* ll = (b128_t*)l;
  __builtin_amdgcn_global_load_async_to_lds_b128((gptr128)gg, (lptr128)ll, 0, 0);
}
#endif

// ---------------- WMMA fragment loaders (ISA 7.12.2 layouts) ----------------
// A 16x32 f16: lanes 0-15 = rows M, halves {0..7}->K+0..7, {8..15}->K+16..23;
// lanes 16-31 same rows, K offset +8.
__device__ __forceinline__ v16h load_a_frag(const half_t* __restrict__ base,
                                            int ld, int mbase, int kbase, int lane) {
  int m = mbase + (lane & 15);
  const half_t* p = base + (size_t)m * ld + kbase + ((lane >> 4) << 3);
  v8h lo = *(const v8h*)p;          // K = k0 .. k0+7
  v8h hi = *(const v8h*)(p + 16);   // K = k0+16 .. k0+23
  v16h a;
#pragma unroll
  for (int i = 0; i < 8; ++i) { a[i] = lo[i]; a[i + 8] = hi[i]; }
  return a;
}

// B 32x16 f16 (NT GEMM: second operand fed as row-major rows):
// lane&15 = column N, halves {0..7}->K+0..7, {8..15}->K+16..23; upper lane half K offset +8.
__device__ __forceinline__ v16h load_b_frag(const half_t* __restrict__ base,
                                            int ld, int nbase, int kbase, int lane) {
  int n = nbase + (lane & 15);
  const half_t* p = base + (size_t)n * ld + kbase + ((lane >> 4) << 3);
  v8h lo = *(const v8h*)p;
  v8h hi = *(const v8h*)(p + 16);
  v16h b;
#pragma unroll
  for (int i = 0; i < 8; ++i) { b[i] = lo[i]; b[i + 8] = hi[i]; }
  return b;
}

// ---------------- small utility kernels ----------------
__global__ __launch_bounds__(256)
void cvt_f32_to_f16_kernel(const float* __restrict__ in, half_t* __restrict__ out, int n) {
  int i = blockIdx.x * 256 + threadIdx.x;
  if (i < n) out[i] = (half_t)in[i];
}

__global__ __launch_bounds__(256)
void init_kernel(const float* __restrict__ src_mask, const float* __restrict__ tgt_mask,
                 float* __restrict__ ms, float* __restrict__ ns,
                 float* __restrict__ u, float* __restrict__ v,
                 float* __restrict__ valid_out) {
  int bs = blockIdx.x;
  int t  = threadIdx.x;
  __shared__ float red[256];

  float cs = 0.f;
  for (int i = t; i < MDIM; i += 256) cs += (src_mask[bs * MDIM + i] > MASK_THR) ? 1.f : 0.f;
  red[t] = cs; __syncthreads();
  for (int o = 128; o > 0; o >>= 1) { if (t < o) red[t] += red[t + o]; __syncthreads(); }
  float msv = red[0]; __syncthreads();

  float ct = 0.f;
  for (int i = t; i < NDIM; i += 256) ct += (tgt_mask[bs * NDIM + i] > MASK_THR) ? 1.f : 0.f;
  red[t] = ct; __syncthreads();
  for (int o = 128; o > 0; o >>= 1) { if (t < o) red[t] += red[t + o]; __syncthreads(); }
  float nsv = red[0];

  for (int i = t; i < UVSTRIDE; i += 256) {
    u[bs * UVSTRIDE + i] = 0.f;
    v[bs * UVSTRIDE + i] = 0.f;
  }
  if (t == 0) {
    ms[bs] = msv;
    ns[bs] = nsv;
    valid_out[bs] = (msv > 0.f && nsv > 0.f) ? 1.f : 0.f;
  }
}

// ---------------- projection GEMM: out = scale * feat @ W^T (f16 out) -------
__global__ __launch_bounds__(32)
void proj_wmma_kernel(const half_t* __restrict__ feat, const half_t* __restrict__ W,
                      half_t* __restrict__ out, float scale) {
  int lane  = threadIdx.x;
  int nbase = blockIdx.x * 16;
  int mbase = blockIdx.y * 16;
  v8f acc = {};
#pragma unroll
  for (int k = 0; k < CDIM; k += 32) {
    v16h a = load_a_frag(feat, CDIM, mbase, k, lane);
    v16h b = load_b_frag(W,    CDIM, nbase, k, lane);   // B[k][n] = W[n][k]
    acc = __builtin_amdgcn_wmma_f32_16x16x32_f16(false, a, false, b,
                                                 (short)0, acc, false, false);
  }
  int mrow = mbase + ((lane >> 4) << 3);
  int ncol = nbase + (lane & 15);
#pragma unroll
  for (int r = 0; r < 8; ++r)
    out[(size_t)(mrow + r) * CDIM + ncol] = (half_t)(acc[r] * scale);
}

// ---------------- sim GEMM + mask + cross write ------------------------------
// Workgroup = 8 waves computes a TILE_M x TILE_N block. Full-K operand tiles
// (A: 128x256 f16, B: 64x256 f16 -- both contiguous global regions) are staged
// into LDS once via GLOBAL_LOAD_ASYNC_TO_LDS_B128 (ASYNCcnt), then each wave
// computes a 32x32 sub-tile (2x2 accumulators, 32 WMMAs, A/B frags reused 2x).
__global__ __launch_bounds__(256)
void sim_wmma_kernel(const half_t* __restrict__ psrc, const half_t* __restrict__ ptgt,
                     const float* __restrict__ src_mask, const float* __restrict__ tgt_mask,
                     float* __restrict__ Z, float* __restrict__ cross_out) {
  extern __shared__ char smem_raw[];
  half_t* sA = (half_t*)smem_raw;                       // TILE_M x 256
  half_t* sB = (half_t*)(smem_raw + SMEM_A_BYTES);      // TILE_N x 256

  int t    = threadIdx.x;
  int lane = t & 31;
  int wave = t >> 5;
  int bs   = blockIdx.z;
  int Mb   = blockIdx.y * TILE_M;
  int Nb   = blockIdx.x * TILE_N;

  const half_t* Ag = psrc + ((size_t)bs * MDIM + Mb) * CDIM;  // contiguous 64 KB
  const half_t* Bg = ptgt + ((size_t)bs * NDIM + Nb) * CDIM;  // contiguous 32 KB

  // ---- stage block tiles into LDS ----
#if HAVE_ASYNC_LDS
#pragma unroll
  for (int i = 0; i < (TILE_M * CDIM) / (8 * 256); ++i) {  // 16 iters: A
    int off = (i * 256 + t) * 8;                           // halves (16B units)
    async_ld_b128(Ag + off, sA + off);
  }
#pragma unroll
  for (int i = 0; i < (TILE_N * CDIM) / (8 * 256); ++i) {  // 8 iters: B
    int off = (i * 256 + t) * 8;
    async_ld_b128(Bg + off, sB + off);
  }
  __builtin_amdgcn_s_wait_asynccnt(0);
  __syncthreads();
#else
#pragma unroll
  for (int i = 0; i < (TILE_M * CDIM) / (8 * 256); ++i) {
    int off = (i * 256 + t) * 8;
    *(v4f*)(sA + off) = *(const v4f*)(Ag + off);
  }
#pragma unroll
  for (int i = 0; i < (TILE_N * CDIM) / (8 * 256); ++i) {
    int off = (i * 256 + t) * 8;
    *(v4f*)(sB + off) = *(const v4f*)(Bg + off);
  }
  __syncthreads();
#endif

  // ---- compute: wave (wm, wn) owns rows [wm*32,+32), cols [wn*32,+32) ----
  int wm = wave & 3;        // 4 waves along M
  int wn = wave >> 2;       // 2 waves along N
  int mrel = wm * 32;
  int nrel = wn * 32;

  v8f acc00 = {}, acc01 = {}, acc10 = {}, acc11 = {};
#pragma unroll
  for (int k = 0; k < CDIM; k += 32) {
    v16h a0 = load_a_frag(sA, CDIM, mrel,      k, lane);
    v16h a1 = load_a_frag(sA, CDIM, mrel + 16, k, lane);
    v16h b0 = load_b_frag(sB, CDIM, nrel,      k, lane);
    v16h b1 = load_b_frag(sB, CDIM, nrel + 16, k, lane);
    acc00 = __builtin_amdgcn_wmma_f32_16x16x32_f16(false, a0, false, b0, (short)0, acc00, false, false);
    acc01 = __builtin_amdgcn_wmma_f32_16x16x32_f16(false, a0, false, b1, (short)0, acc01, false, false);
    acc10 = __builtin_amdgcn_wmma_f32_16x16x32_f16(false, a1, false, b0, (short)0, acc10, false, false);
    acc11 = __builtin_amdgcn_wmma_f32_16x16x32_f16(false, a1, false, b1, (short)0, acc11, false, false);
  }

  // ---- epilogue: mask to -inf, write Z and cross ----
  const float* sm = src_mask + (size_t)bs * MDIM;
  const float* tm = tgt_mask + (size_t)bs * NDIM;
  float* Zs = Z         + (size_t)bs * MDIM * NDIM;
  float* Cs = cross_out + (size_t)bs * MDIM * NDIM;
  int lrow = (lane >> 4) << 3;   // 0 or 8
  int lcol = lane & 15;

  v8f accs[2][2] = { { acc00, acc01 }, { acc10, acc11 } };
#pragma unroll
  for (int ni = 0; ni < 2; ++ni) {
    int n = Nb + nrel + ni * 16 + lcol;
    bool tb = tm[n] > MASK_THR;
#pragma unroll
    for (int mi = 0; mi < 2; ++mi) {
      int mbaseg = Mb + mrel + mi * 16 + lrow;
#pragma unroll
      for (int r = 0; r < 8; ++r) {
        int m = mbaseg + r;
        bool c = (sm[m] > MASK_THR) && tb;
        size_t idx = (size_t)m * NDIM + n;
        Zs[idx] = c ? accs[mi][ni][r] : NEG_INF;
        Cs[idx] = c ? 1.0f : 0.0f;
      }
    }
  }
}

// ---------------- -inf-safe streaming logsumexp ------------------------------
__device__ __forceinline__ void lse_push(float x, float& mx, float& s) {
  if (x > mx)               { s = s * __expf(mx - x) + 1.0f; mx = x; }
  else if (mx > NEG_INF)    { s += __expf(x - mx); }
}
__device__ __forceinline__ void lse_merge(float& mx, float& s, float m2, float s2) {
  if (m2 > mx) { float tm = mx; mx = m2; m2 = tm; float ts = s; s = s2; s2 = ts; }
  if (mx > NEG_INF) s += s2 * __expf(m2 - mx);
}
__device__ __forceinline__ float lse_wave(float mx, float s) {
#pragma unroll
  for (int off = 16; off > 0; off >>= 1) {
    float m2 = __shfl_xor(mx, off, 32);
    float s2 = __shfl_xor(s,  off, 32);
    lse_merge(mx, s, m2, s2);
  }
  return (mx > NEG_INF) ? (mx + __logf(s)) : NEG_INF;
}

// ---------------- Sinkhorn: u update (one wave per row, incl. bin row) -------
__global__ __launch_bounds__(32)
void u_update_kernel(const float* __restrict__ Z, const float* __restrict__ v,
                     float* __restrict__ u, const float* __restrict__ ms,
                     const float* __restrict__ ns, const float* __restrict__ alpha_p) {
  int bs = blockIdx.y, i = blockIdx.x, lane = threadIdx.x;
  float alpha = *alpha_p;
  float msv = ms[bs], nsv = ns[bs];
  float norm = -__logf(msv + nsv);
  const float* vv = v + bs * UVSTRIDE;
  float mx = NEG_INF, s = 0.f;
  if (i < MDIM) {
    const float* zr = Z + ((size_t)bs * MDIM + i) * NDIM;
    for (int j = lane; j < NDIM; j += 32) lse_push(zr[j] + vv[j], mx, s);
    if (lane == 0) lse_push(alpha + vv[NDIM], mx, s);   // bin column term
    float lse = lse_wave(mx, s);
    if (lane == 0) u[bs * UVSTRIDE + i] = norm - lse;
  } else {  // bin row: LSE_j(alpha + v[j]), j = 0..N
    for (int j = lane; j <= NDIM; j += 32) lse_push(alpha + vv[j], mx, s);
    float lse = lse_wave(mx, s);
    if (lane == 0) u[bs * UVSTRIDE + MDIM] = __logf(nsv) + norm - lse;
  }
}

// ---------------- Sinkhorn: v update (thread per column, coalesced) ----------
__global__ __launch_bounds__(256)
void v_update_kernel(const float* __restrict__ Z, const float* __restrict__ u,
                     float* __restrict__ v, const float* __restrict__ ms,
                     const float* __restrict__ ns, const float* __restrict__ alpha_p) {
  int bs = blockIdx.y;
  float alpha = *alpha_p;
  float msv = ms[bs], nsv = ns[bs];
  float norm = -__logf(msv + nsv);
  const float* uu = u + bs * UVSTRIDE;
  __shared__ float shm[256], shs[256];

  if (blockIdx.x < NDIM / 256) {
    int j = blockIdx.x * 256 + threadIdx.x;
    const float* zc = Z + (size_t)bs * MDIM * NDIM;
    float mx = NEG_INF, s = 0.f;
    for (int i = 0; i < MDIM; ++i) lse_push(zc[(size_t)i * NDIM + j] + uu[i], mx, s);
    lse_push(alpha + uu[MDIM], mx, s);                  // bin row term
    float lse = (mx > NEG_INF) ? (mx + __logf(s)) : NEG_INF;
    v[bs * UVSTRIDE + j] = norm - lse;
  } else {  // bin column: LSE_i(alpha + u[i]), i = 0..M, block-wide reduce
    float mx = NEG_INF, s = 0.f;
    for (int i = threadIdx.x; i <= MDIM; i += 256) lse_push(alpha + uu[i], mx, s);
    shm[threadIdx.x] = mx; shs[threadIdx.x] = s; __syncthreads();
    for (int o = 128; o > 0; o >>= 1) {
      if (threadIdx.x < o) {
        float m1 = shm[threadIdx.x], s1 = shs[threadIdx.x];
        lse_merge(m1, s1, shm[threadIdx.x + o], shs[threadIdx.x + o]);
        shm[threadIdx.x] = m1; shs[threadIdx.x] = s1;
      }
      __syncthreads();
    }
    if (threadIdx.x == 0) {
      float lse = (shm[0] > NEG_INF) ? (shm[0] + __logf(shs[0])) : NEG_INF;
      v[bs * UVSTRIDE + NDIM] = __logf(msv) + norm - lse;
    }
  }
}

// ---------------- conf = exp(Z+u+v-norm) in place, + row max ----------------
__global__ __launch_bounds__(32)
void conf_rowmax_kernel(float* __restrict__ Z, const float* __restrict__ u,
                        const float* __restrict__ v, const float* __restrict__ ms,
                        const float* __restrict__ ns, float* __restrict__ rowmax) {
  int bs = blockIdx.y, i = blockIdx.x, lane = threadIdx.x;
  float norm = -__logf(ms[bs] + ns[bs]);
  float ui = u[bs * UVSTRIDE + i];
  const float* vv = v + bs * UVSTRIDE;
  float* zr = Z + ((size_t)bs * MDIM + i) * NDIM;
  float mx = 0.f;
  for (int j = lane; j < NDIM; j += 32) {
    float c = __expf(zr[j] + ui + vv[j] - norm);
    zr[j] = c;
    mx = fmaxf(mx, c);
  }
#pragma unroll
  for (int off = 16; off > 0; off >>= 1) mx = fmaxf(mx, __shfl_xor(mx, off, 32));
  if (lane == 0) rowmax[bs * MDIM + i] = mx;
}

__global__ __launch_bounds__(256)
void colmax_kernel(const float* __restrict__ conf, float* __restrict__ colmax) {
  int bs = blockIdx.y;
  int j = blockIdx.x * 256 + threadIdx.x;
  const float* c = conf + (size_t)bs * MDIM * NDIM;
  float mx = 0.f;
  for (int i = 0; i < MDIM; ++i) mx = fmaxf(mx, c[(size_t)i * NDIM + j]);
  colmax[bs * NDIM + j] = mx;
}

__global__ __launch_bounds__(256)
void mask_kernel(const float* __restrict__ conf, const float* __restrict__ rowmax,
                 const float* __restrict__ colmax, float* __restrict__ mask_out,
                 float* __restrict__ mconf_out) {
  size_t idx = (size_t)blockIdx.x * 256 + threadIdx.x;  // covers BS*M*N exactly
  size_t j   = idx & (NDIM - 1);
  size_t row = idx >> 11;            // bs*M + i
  size_t bs  = row >> 11;
  float c = conf[idx];
  bool m = (c > CONF_THR) && (c == rowmax[row]) && (c == colmax[(bs << 11) + j]);
  mask_out[idx]  = m ? 1.0f : 0.0f;
  mconf_out[idx] = m ? c    : 0.0f;
}

// ---------------------------------------------------------------------------
extern "C" void kernel_launch(void* const* d_in, const int* in_sizes, int n_in,
                              void* d_out, int out_size, void* d_ws, size_t ws_size,
                              hipStream_t stream) {
  (void)in_sizes; (void)n_in; (void)out_size; (void)ws_size;
  const float* src_feats = (const float*)d_in[0];   // (8, 2048, 256)
  const float* tgt_feats = (const float*)d_in[1];   // (8, 2048, 256)
  const float* src_mask  = (const float*)d_in[2];   // (8, 2048)
  const float* tgt_mask  = (const float*)d_in[3];   // (8, 2048)
  const float* W_src     = (const float*)d_in[4];   // (256, 256)
  const float* alpha_p   = (const float*)d_in[5];   // scalar

  const size_t PLANE = (size_t)BSLICES * MDIM * NDIM;   // 33554432
  float* o_conf  = (float*)d_out;
  float* o_mask  = o_conf  + PLANE;
  float* o_mconf = o_mask  + PLANE;
  float* o_cross = o_mconf + PLANE;
  float* o_valid = o_cross + PLANE;

  // workspace layout: f16 region then f32 region
  const int FEATS = BSLICES * MDIM * CDIM;              // 4194304
  half_t* h_src  = (half_t*)d_ws;
  half_t* h_tgt  = h_src  + FEATS;
  half_t* h_W    = h_tgt  + FEATS;                      // 65536
  half_t* h_psrc = h_W    + CDIM * CDIM;
  half_t* h_ptgt = h_psrc + FEATS;
  float*  f_u      = (float*)(h_ptgt + FEATS);
  float*  f_v      = f_u      + BSLICES * UVSTRIDE;
  float*  f_ms     = f_v      + BSLICES * UVSTRIDE;
  float*  f_ns     = f_ms     + BSLICES;
  float*  f_rowmax = f_ns     + BSLICES;
  float*  f_colmax = f_rowmax + BSLICES * MDIM;

  float* Z = o_conf;   // masked similarity lives in conf output region; exp'd in place

  const float scale = 1.0f / 16.0f;   // 1/sqrt(256)

  // 1. convert inputs to f16
  cvt_f32_to_f16_kernel<<<(FEATS + 255) / 256, 256, 0, stream>>>(src_feats, h_src, FEATS);
  cvt_f32_to_f16_kernel<<<(FEATS + 255) / 256, 256, 0, stream>>>(tgt_feats, h_tgt, FEATS);
  cvt_f32_to_f16_kernel<<<(CDIM * CDIM + 255) / 256, 256, 0, stream>>>(W_src, h_W, CDIM * CDIM);

  // 2. ms/ns/valid + zero u,v
  init_kernel<<<BSLICES, 256, 0, stream>>>(src_mask, tgt_mask, f_ms, f_ns, f_u, f_v, o_valid);

  // 3. projection (WMMA): 16x16 tile per wave
  {
    dim3 grid(CDIM / 16, (BSLICES * MDIM) / 16);
    proj_wmma_kernel<<<grid, 32, 0, stream>>>(h_src, h_W, h_psrc, scale);
    proj_wmma_kernel<<<grid, 32, 0, stream>>>(h_tgt, h_W, h_ptgt, scale);
  }

  // 4. similarity (WMMA, LDS macro-tile) + cross mask
  {
    dim3 grid(NDIM / TILE_N, MDIM / TILE_M, BSLICES);
    sim_wmma_kernel<<<grid, 256, SMEM_BYTES, stream>>>(h_psrc, h_ptgt, src_mask,
                                                       tgt_mask, Z, o_cross);
  }

  // 5. Sinkhorn (3 iterations)
  for (int it = 0; it < SKH_ITERS; ++it) {
    u_update_kernel<<<dim3(MDIM + 1, BSLICES), 32, 0, stream>>>(Z, f_v, f_u, f_ms, f_ns, alpha_p);
    v_update_kernel<<<dim3(NDIM / 256 + 1, BSLICES), 256, 0, stream>>>(Z, f_u, f_v, f_ms, f_ns, alpha_p);
  }

  // 6. conf = exp(logZ) in place + row max
  conf_rowmax_kernel<<<dim3(MDIM, BSLICES), 32, 0, stream>>>(Z, f_u, f_v, f_ms, f_ns, f_rowmax);

  // 7. col max
  colmax_kernel<<<dim3(NDIM / 256, BSLICES), 256, 0, stream>>>(o_conf, f_colmax);

  // 8. mutual-NN mask + mconf
  mask_kernel<<<(unsigned)(PLANE / 256), 256, 0, stream>>>(o_conf, f_rowmax, f_colmax, o_mask, o_mconf);
}
